// MochaAttention_45079976739212
// MI455X (gfx1250) — matrix operand order repaired and checked by reading, not compiled, not run on hardware
//
#include <hip/hip_runtime.h>
#include <hip/hip_bf16.h>
#include <math.h>

#define T_LEN 600
#define S_LEN 32
#define CELL_DIM 1024
#define SYM_LEN 48
#define NCLS 5000
#define WIN_W 8
#define CLIP_V 50.0f
#define EPS_V 1e-10f
#define NEG_V -1e30f
#define KCAT 1536 /* emb(512) + ctx(512) + r(512) */

typedef __attribute__((ext_vector_type(16))) __bf16 v16bf;
typedef __attribute__((ext_vector_type(8)))  __bf16 v8bf;
typedef __attribute__((ext_vector_type(8)))  float  v8f;

static __device__ __forceinline__ v8f wmma_bf16(v16bf a, v16bf b, v8f c) {
  return __builtin_amdgcn_wmma_f32_16x16x32_bf16(false, a, false, b, (short)0, c, false, false);
}

// A fragment: 16x32 bf16 tile from row-major [M,K]; lane(l&15)=row, half=(l>>4):
// per-lane data = two contiguous 16B chunks at K = 8h and K = 16+8h.
static __device__ __forceinline__ v16bf load_a_frag(const __bf16* src, int ld, int m0, int k0, int lane) {
  int m = m0 + (lane & 15);
  int half = lane >> 4;
  const __bf16* p = src + (long)m * ld + k0 + 8 * half;
  v16bf a;
#pragma unroll
  for (int i = 0; i < 8; ++i) { a[i] = p[i]; a[i + 8] = p[16 + i]; }
  return a;
}

// B fragment: 32x16 tile with B supplied as W[N,K] row-major (X @ W.T GEMMs):
// lane(l&15)=col n; two contiguous 16B chunks at W + n*K + k0 + 8*half and + 16.
static __device__ __forceinline__ v16bf load_b_frag(const __bf16* w, int ld, int n0, int k0, int lane) {
  int n = n0 + (lane & 15);
  int half = lane >> 4;
  const __bf16* p = w + (long)n * ld + k0 + 8 * half;
  v16bf b;
#pragma unroll
  for (int i = 0; i < 8; ++i) { b[i] = p[i]; b[i + 8] = p[16 + i]; }
  return b;
}

static __device__ __forceinline__ void store_c_f32(float* dst, int ld, int m0, int n0, v8f c, int lane) {
  int n = n0 + (lane & 15);
  int rb = m0 + 8 * (lane >> 4);
#pragma unroll
  for (int v = 0; v < 8; ++v) dst[(long)(rb + v) * ld + n] = c[v];
}
static __device__ __forceinline__ void store_c_bf16(__bf16* dst, int ld, int m0, int n0, v8f c, int lane) {
  int n = n0 + (lane & 15);
  int rb = m0 + 8 * (lane >> 4);
#pragma unroll
  for (int v = 0; v < 8; ++v) dst[(long)(rb + v) * ld + n] = (__bf16)c[v];
}

/* software-pipelined 32xN GEMM body: 2 M-tiles x 4 N-tiles, K = KTOT.
   Next iteration's fragments are fetched before the current WMMAs issue, so
   VMEM latency overlaps matrix issue instead of s_wait_loadcnt 0 per WMMA. */
template <int KTOT>
static __device__ __forceinline__ void gemm_2x4(const __bf16* A, int lda, const __bf16* W, int ldw,
                                                int n0, int lane, v8f acc[2][4]) {
  v16bf a0 = load_a_frag(A, lda, 0, 0, lane);
  v16bf a1 = load_a_frag(A, lda, 16, 0, lane);
  v16bf b0 = load_b_frag(W, ldw, n0, 0, lane);
  v16bf b1 = load_b_frag(W, ldw, n0 + 16, 0, lane);
  v16bf b2 = load_b_frag(W, ldw, n0 + 32, 0, lane);
  v16bf b3 = load_b_frag(W, ldw, n0 + 48, 0, lane);
  for (int k0 = 32; k0 < KTOT; k0 += 32) {
    v16bf na0 = load_a_frag(A, lda, 0, k0, lane);
    v16bf na1 = load_a_frag(A, lda, 16, k0, lane);
    v16bf nb0 = load_b_frag(W, ldw, n0, k0, lane);
    v16bf nb1 = load_b_frag(W, ldw, n0 + 16, k0, lane);
    v16bf nb2 = load_b_frag(W, ldw, n0 + 32, k0, lane);
    v16bf nb3 = load_b_frag(W, ldw, n0 + 48, k0, lane);
    acc[0][0] = wmma_bf16(a0, b0, acc[0][0]);
    acc[1][0] = wmma_bf16(a1, b0, acc[1][0]);
    acc[0][1] = wmma_bf16(a0, b1, acc[0][1]);
    acc[1][1] = wmma_bf16(a1, b1, acc[1][1]);
    acc[0][2] = wmma_bf16(a0, b2, acc[0][2]);
    acc[1][2] = wmma_bf16(a1, b2, acc[1][2]);
    acc[0][3] = wmma_bf16(a0, b3, acc[0][3]);
    acc[1][3] = wmma_bf16(a1, b3, acc[1][3]);
    a0 = na0; a1 = na1; b0 = nb0; b1 = nb1; b2 = nb2; b3 = nb3;
  }
  acc[0][0] = wmma_bf16(a0, b0, acc[0][0]);
  acc[1][0] = wmma_bf16(a1, b0, acc[1][0]);
  acc[0][1] = wmma_bf16(a0, b1, acc[0][1]);
  acc[1][1] = wmma_bf16(a1, b1, acc[1][1]);
  acc[0][2] = wmma_bf16(a0, b2, acc[0][2]);
  acc[1][2] = wmma_bf16(a1, b2, acc[1][2]);
  acc[0][3] = wmma_bf16(a0, b3, acc[0][3]);
  acc[1][3] = wmma_bf16(a1, b3, acc[1][3]);
}

/* pipelined 16xN strip: 1 M-tile x 4 N-tiles */
template <int KTOT>
static __device__ __forceinline__ void gemm_1x4(const __bf16* A, int lda, int m0, const __bf16* W,
                                                int ldw, int n0, int lane, v8f acc[4]) {
  v16bf a = load_a_frag(A, lda, m0, 0, lane);
  v16bf b0 = load_b_frag(W, ldw, n0, 0, lane);
  v16bf b1 = load_b_frag(W, ldw, n0 + 16, 0, lane);
  v16bf b2 = load_b_frag(W, ldw, n0 + 32, 0, lane);
  v16bf b3 = load_b_frag(W, ldw, n0 + 48, 0, lane);
  for (int k0 = 32; k0 < KTOT; k0 += 32) {
    v16bf na = load_a_frag(A, lda, m0, k0, lane);
    v16bf nb0 = load_b_frag(W, ldw, n0, k0, lane);
    v16bf nb1 = load_b_frag(W, ldw, n0 + 16, k0, lane);
    v16bf nb2 = load_b_frag(W, ldw, n0 + 32, k0, lane);
    v16bf nb3 = load_b_frag(W, ldw, n0 + 48, k0, lane);
    acc[0] = wmma_bf16(a, b0, acc[0]);
    acc[1] = wmma_bf16(a, b1, acc[1]);
    acc[2] = wmma_bf16(a, b2, acc[2]);
    acc[3] = wmma_bf16(a, b3, acc[3]);
    a = na; b0 = nb0; b1 = nb1; b2 = nb2; b3 = nb3;
  }
  acc[0] = wmma_bf16(a, b0, acc[0]);
  acc[1] = wmma_bf16(a, b1, acc[1]);
  acc[2] = wmma_bf16(a, b2, acc[2]);
  acc[3] = wmma_bf16(a, b3, acc[3]);
}

static __device__ __forceinline__ float sigmf(float x) { return 1.f / (1.f + expf(-x)); }

static __device__ __forceinline__ float wave_incl_scan(float x, int lane) {
#pragma unroll
  for (int off = 1; off < 32; off <<= 1) {
    float y = __shfl_up(x, off, 32);
    if (lane >= off) x += y;
  }
  return x;
}

/* ---------------- precompute ---------------- */

__global__ void k_build_wcat(const float* Wx, const float* Wr, __bf16* Wcat) {
  long total = 4096L * KCAT;
  for (long i = (long)blockIdx.x * blockDim.x + threadIdx.x; i < total;
       i += (long)gridDim.x * blockDim.x) {
    long n = i / KCAT, k = i % KCAT;
    float v = (k < 1024) ? Wx[n * 1024 + k] : Wr[n * 512 + (k - 1024)];
    Wcat[i] = (__bf16)v;
  }
}

__global__ void k_f2bf(const float* src, __bf16* dst, long n) {
  for (long i = (long)blockIdx.x * blockDim.x + threadIdx.x; i < n;
       i += (long)gridDim.x * blockDim.x)
    dst[i] = (__bf16)src[i];
}

__global__ void k_cat2(const float* A, const float* B, __bf16* dst, long n_each) {
  for (long i = (long)blockIdx.x * blockDim.x + threadIdx.x; i < 2 * n_each;
       i += (long)gridDim.x * blockDim.x)
    dst[i] = (__bf16)((i < n_each) ? A[i] : B[i - n_each]);
}

/* ahch[t*32+s, 0:512]=data@W_at_h.T, [512:1024]=data@W_ch_h.T ; M=19200,K=512,N=1024 */
__global__ void __launch_bounds__(256) k_gemm_ahch(const __bf16* A, const __bf16* W, __bf16* Cb) {
  int lane = threadIdx.x & 31;
  int gw = blockIdx.x * (blockDim.x >> 5) + (threadIdx.x >> 5);
  int m0 = (gw >> 4) * 16;
  int n0 = (gw & 15) * 64;
  v8f acc[4] = {};
  gemm_1x4<512>(A, 512, m0, W, 512, n0, lane, acc);
#pragma unroll
  for (int j = 0; j < 4; ++j) store_c_bf16(Cb, 1024, m0, n0 + 16 * j, acc[j], lane);
}

__global__ void k_init(__bf16* xcat, float* cstate, float* alphaA) {
  int i = blockIdx.x * blockDim.x + threadIdx.x;
  int stride = gridDim.x * blockDim.x;
  for (int j = i; j < 32 * KCAT; j += stride) xcat[j] = (__bf16)0.f;
  for (int j = i; j < 32 * CELL_DIM; j += stride) cstate[j] = 0.f;
  for (int j = i; j < 32 * T_LEN; j += stride) alphaA[j] = (j % T_LEN == 0) ? 1.f : 0.f;
}

/* ---------------- per-step kernels ---------------- */

__global__ void k_embed(const int* att_label, const float* E, __bf16* xcat, int sym) {
  int i = blockIdx.x * blockDim.x + threadIdx.x;
  if (i >= 32 * 512) return;
  int s = i >> 9, m = i & 511;
  int lab = (sym == 0) ? (NCLS - 2) : att_label[(sym - 1) * S_LEN + s];
  float v = (lab >= 0) ? E[(long)lab * 512 + m] : 0.f;
  xcat[s * KCAT + m] = (__bf16)v;
}

/* fused gates GEMM + LSTM cell. 16 blocks x 128 thr; wave = gate. */
__global__ void __launch_bounds__(128) k_gates(const __bf16* xcat, const __bf16* Wcat,
                                               const float* bias, float* cstate, __bf16* hbf) {
  __shared__ float lds[4][32][64];
  int lane = threadIdx.x & 31;
  int g = threadIdx.x >> 5;
  int n0c = blockIdx.x * 64;
  int ncol = g * 1024 + n0c;
  v8f acc[2][4] = {};
  gemm_2x4<KCAT>(xcat, KCAT, Wcat, KCAT, ncol, lane, acc);
#pragma unroll
  for (int mt = 0; mt < 2; ++mt)
#pragma unroll
    for (int j = 0; j < 4; ++j)
      store_c_f32(&lds[g][0][0], 64, mt * 16, j * 16, acc[mt][j], lane);
  __syncthreads();
#pragma unroll 4
  for (int e = 0; e < 16; ++e) {
    int flat = threadIdx.x + 128 * e;
    int s = flat >> 6, col = flat & 63;
    int cell = n0c + col;
    float gi = lds[0][s][col] + bias[cell];
    float gf = lds[1][s][col] + bias[1024 + cell];
    float gg = lds[2][s][col] + bias[2048 + cell];
    float go = lds[3][s][col] + bias[3072 + cell];
    float c_old = cstate[s * CELL_DIM + cell];
    float cn = sigmf(gf) * c_old + sigmf(gi) * tanhf(gg);
    cn = fminf(fmaxf(cn, -CLIP_V), CLIP_V);
    cstate[s * CELL_DIM + cell] = cn;
    hbf[s * CELL_DIM + cell] = (__bf16)(sigmf(go) * tanhf(cn));
  }
}

/* r = h @ Wp.T ; bf16 into xcat[:,1024:], masked f32 into out */
__global__ void __launch_bounds__(32) k_proj(const __bf16* hbf, const __bf16* Wpb,
                                             const float* att_mask, __bf16* xcat,
                                             float* outp, int sym) {
  int lane = threadIdx.x & 31;
  int n0 = blockIdx.x * 64;
  v8f acc[2][4] = {};
  gemm_2x4<1024>(hbf, 1024, Wpb, 1024, n0, lane, acc);
#pragma unroll
  for (int mt = 0; mt < 2; ++mt)
#pragma unroll
    for (int j = 0; j < 4; ++j) {
      int n = n0 + 16 * j + (lane & 15);
      int rb = mt * 16 + 8 * (lane >> 4);
#pragma unroll
      for (int v = 0; v < 8; ++v) {
        int row = rb + v;
        float val = acc[mt][j][v];
        xcat[row * KCAT + 1024 + n] = (__bf16)val;
        float am = att_mask[sym * S_LEN + row];
        outp[((long)sym * 32 + row) * 1024 + n] = val * am;
      }
    }
}

/* st_cat[s,0:512]=r@W_at_s.T, [512:1024]=r@W_ch_s.T */
__global__ void __launch_bounds__(32) k_st(const __bf16* xcat, const __bf16* Wscat, float* st_cat) {
  int lane = threadIdx.x & 31;
  int n0 = blockIdx.x * 64;
  v8f acc[2][4] = {};
  gemm_2x4<512>(xcat + 1024, KCAT, Wscat, 512, n0, lane, acc);
#pragma unroll
  for (int mt = 0; mt < 2; ++mt)
#pragma unroll
    for (int j = 0; j < 4; ++j)
      store_c_f32(st_cat, 1024, mt * 16, n0 + 16 * j, acc[mt][j], lane);
}

/* energies: e[s,t] = sum_m v[m]*tanh(st[s,m]+ah[t,s,m]); lane owns a contiguous
   16-wide chunk of m so ahch reads are two 16B vector loads per row. */
__global__ void __launch_bounds__(256) k_energy(const float* st_cat, const __bf16* ahch,
                                                const float* v_at, const float* v_ch,
                                                const float* rnn_mask, float* eT, float* ueT) {
  int t = blockIdx.x;
  int lane = threadIdx.x & 31;
  int w = threadIdx.x >> 5;
  int tsel = w >> 2;
  int sbase = (w & 3) * 8;
  const float* vv = (tsel ? v_ch : v_at) + lane * 16;
  float* dst = tsel ? ueT : eT;
  float vreg[16];
#pragma unroll
  for (int i = 0; i < 16; ++i) vreg[i] = vv[i];
  for (int q = 0; q < 8; ++q) {
    int s = sbase + q;
    const float* st16 = st_cat + s * 1024 + tsel * 512 + lane * 16;
    const v8bf* ah8 = (const v8bf*)(ahch + ((long)t * 32 + s) * 1024 + tsel * 512 + lane * 16);
    v8bf A0 = ah8[0], A1 = ah8[1];
    float sum = 0.f;
#pragma unroll
    for (int i = 0; i < 8; ++i) sum += vreg[i] * tanhf(st16[i] + (float)A0[i]);
#pragma unroll
    for (int i = 0; i < 8; ++i) sum += vreg[8 + i] * tanhf(st16[8 + i] + (float)A1[i]);
#pragma unroll
    for (int off = 16; off > 0; off >>= 1) sum += __shfl_xor(sum, off, 32);
    if (lane == 0) {
      float msk = rnn_mask[t * 32 + s];
      dst[s * T_LEN + t] = (msk > 0.f) ? sum : NEG_V;
    }
  }
}

/* per-row scans: alpha recurrence + windowed beta; wave w owns row s=w */
__global__ void __launch_bounds__(1024) k_scan(const float* eT, const float* ueT,
                                               const float* alphaIn, float* alphaOut,
                                               float* csbuf, float* cs2buf, float* betaT) {
  int lane = threadIdx.x & 31;
  int s = threadIdx.x >> 5;
  const float* eRow = eT + s * T_LEN;
  const float* uRow = ueT + s * T_LEN;

  float m = -3.4e38f;
  for (int t = lane; t < T_LEN; t += 32) m = fmaxf(m, uRow[t]);
#pragma unroll
  for (int off = 16; off > 0; off >>= 1) m = fmaxf(m, __shfl_xor(m, off, 32));
  float umax = m;

  float carry_lq = 0.f, carry_af = 0.f;
  for (int t0 = 0; t0 < T_LEN; t0 += 32) {
    int t = t0 + lane;
    bool inb = t < T_LEN;
    float e = inb ? eRow[t] : NEG_V;
    float p = sigmf(e);
    float lq = inb ? logf(fmaxf(1.f - p, EPS_V)) : 0.f;
    float ilq = wave_incl_scan(lq, lane);
    float cp = expf(carry_lq + ilq - lq);
    float af = inb ? alphaIn[s * T_LEN + t] / fmaxf(cp, EPS_V) : 0.f;
    float iaf = wave_incl_scan(af, lane);
    float an = p * cp * (carry_af + iaf);
    if (inb) alphaOut[s * T_LEN + t] = an;
    carry_lq += __shfl(ilq, 31, 32);
    carry_af += __shfl(iaf, 31, 32);
  }
  float carry_eu = 0.f;
  for (int t0 = 0; t0 < T_LEN; t0 += 32) {
    int t = t0 + lane;
    bool inb = t < T_LEN;
    float eu = inb ? expf(uRow[t] - umax) : 0.f;
    float ieu = wave_incl_scan(eu, lane);
    if (inb) csbuf[s * 608 + t] = carry_eu + ieu;
    carry_eu += __shfl(ieu, 31, 32);
  }
  float carry_fr = 0.f;
  for (int t0 = 0; t0 < T_LEN; t0 += 32) {
    int t = t0 + lane;
    bool inb = t < T_LEN;
    float fr = 0.f;
    if (inb) {
      float cs = csbuf[s * 608 + t];
      float csm = (t >= WIN_W) ? csbuf[s * 608 + t - WIN_W] : 0.f;
      fr = alphaOut[s * T_LEN + t] / fmaxf(cs - csm, EPS_V);
    }
    float ifr = wave_incl_scan(fr, lane);
    if (inb) cs2buf[s * 608 + t] = carry_fr + ifr;
    carry_fr += __shfl(ifr, 31, 32);
  }
  for (int t0 = 0; t0 < T_LEN; t0 += 32) {
    int t = t0 + lane;
    if (t < T_LEN) {
      int tu = t + WIN_W - 1; if (tu > T_LEN - 1) tu = T_LEN - 1;
      float upper = cs2buf[s * 608 + tu];
      float lower = (t >= 1) ? cs2buf[s * 608 + t - 1] : 0.f;
      float eu = expf(uRow[t] - umax);
      betaT[s * T_LEN + t] = eu * (upper - lower);
    }
  }
}

/* ctx[s,d] = sum_t beta[s,t]*data[t,s,d]; masked out write + bf16 into xcat */
__global__ void __launch_bounds__(256) k_ctx(const float* data, const float* betaT,
                                             const float* att_mask, __bf16* xcat,
                                             float* outp, int sym) {
  int s = blockIdx.x;
  int d0 = threadIdx.x, d1 = threadIdx.x + 256;
  float a0 = 0.f, a1 = 0.f;
  const float* brow = betaT + s * T_LEN;
  for (int t = 0; t < T_LEN; ++t) {
    float bt = brow[t];
    const float* dr = data + ((long)t * 32 + s) * 512;
    __builtin_prefetch(dr + 512 * 32, 0, 0);
    a0 = fmaf(bt, dr[d0], a0);
    a1 = fmaf(bt, dr[d1], a1);
  }
  float am = att_mask[sym * 32 + s];
  xcat[s * KCAT + 512 + d0] = (__bf16)a0;
  xcat[s * KCAT + 512 + d1] = (__bf16)a1;
  long ob = ((long)sym * 32 + s) * 1024 + 512;
  outp[ob + d0] = a0 * am;
  outp[ob + d1] = a1 * am;
}

/* ---------------- host ---------------- */

extern "C" void kernel_launch(void* const* d_in, const int* in_sizes, int n_in,
                              void* d_out, int out_size, void* d_ws, size_t ws_size,
                              hipStream_t stream) {
  (void)in_sizes; (void)n_in; (void)out_size; (void)ws_size;
  const float* data     = (const float*)d_in[0];
  const float* rnn_mask = (const float*)d_in[1];
  const int*   att_label= (const int*)d_in[2];
  const float* att_mask = (const float*)d_in[3];
  const float* E        = (const float*)d_in[4];
  const float* Wx       = (const float*)d_in[5];
  const float* Wr       = (const float*)d_in[6];
  const float* bias     = (const float*)d_in[7];
  const float* Wp       = (const float*)d_in[8];
  const float* W_at_s   = (const float*)d_in[9];
  const float* W_at_h   = (const float*)d_in[10];
  const float* v_at     = (const float*)d_in[11];
  const float* W_ch_s   = (const float*)d_in[12];
  const float* W_ch_h   = (const float*)d_in[13];
  const float* v_ch     = (const float*)d_in[14];
  float* outp = (float*)d_out;

  char* ws = (char*)d_ws;
  size_t off = 0;
  auto alloc = [&](size_t bytes) -> char* {
    char* p = ws + off;
    off += (bytes + 255) & ~(size_t)255;
    return p;
  };
  __bf16* Wcat   = (__bf16*)alloc(4096L * KCAT * 2);
  __bf16* Wpb    = (__bf16*)alloc(512L * 1024 * 2);
  __bf16* Wscat  = (__bf16*)alloc(1024L * 512 * 2);
  __bf16* Whcat  = (__bf16*)alloc(1024L * 512 * 2);
  __bf16* dataBf = (__bf16*)alloc(19200L * 512 * 2);
  __bf16* ahch   = (__bf16*)alloc(19200L * 1024 * 2);
  __bf16* xcat   = (__bf16*)alloc(32L * KCAT * 2);
  float*  cstate = (float*)alloc(32L * CELL_DIM * 4);
  __bf16* hbf    = (__bf16*)alloc(32L * CELL_DIM * 2);
  float*  st_cat = (float*)alloc(32L * 1024 * 4);
  float*  eT     = (float*)alloc(32L * T_LEN * 4);
  float*  ueT    = (float*)alloc(32L * T_LEN * 4);
  float*  alphaA = (float*)alloc(32L * T_LEN * 4);
  float*  alphaB = (float*)alloc(32L * T_LEN * 4);
  float*  csbuf  = (float*)alloc(32L * 608 * 4);
  float*  cs2buf = (float*)alloc(32L * 608 * 4);
  float*  betaT  = (float*)alloc(32L * T_LEN * 4);

  k_build_wcat<<<512, 256, 0, stream>>>(Wx, Wr, Wcat);
  k_f2bf<<<512, 256, 0, stream>>>(Wp, Wpb, 512L * 1024);
  k_cat2<<<512, 256, 0, stream>>>(W_at_s, W_ch_s, Wscat, 512L * 512);
  k_cat2<<<512, 256, 0, stream>>>(W_at_h, W_ch_h, Whcat, 512L * 512);
  k_f2bf<<<1024, 256, 0, stream>>>(data, dataBf, 19200L * 512);
  k_gemm_ahch<<<2400, 256, 0, stream>>>(dataBf, Whcat, ahch);
  k_init<<<256, 256, 0, stream>>>(xcat, cstate, alphaA);

  for (int sym = 0; sym < SYM_LEN; ++sym) {
    float* aIn  = (sym & 1) ? alphaB : alphaA;
    float* aOut = (sym & 1) ? alphaA : alphaB;
    k_embed<<<64, 256, 0, stream>>>(att_label, E, xcat, sym);
    k_gates<<<16, 128, 0, stream>>>(xcat, Wcat, bias, cstate, hbf);
    k_proj<<<8, 32, 0, stream>>>(hbf, Wpb, att_mask, xcat, outp, sym);
    k_st<<<16, 32, 0, stream>>>(xcat, Wscat, st_cat);
    k_energy<<<600, 256, 0, stream>>>(st_cat, ahch, v_at, v_ch, rnn_mask, eT, ueT);
    k_scan<<<1, 1024, 0, stream>>>(eT, ueT, aIn, aOut, csbuf, cs2buf, betaT);
    k_ctx<<<32, 256, 0, stream>>>(data, betaT, att_mask, xcat, outp, sym);
  }
}
